// EdgeFeatCat_75952201663100
// MI455X (gfx1250) — matrix operand order repaired and checked by reading, not compiled, not run on hardware
//
#include <hip/hip_runtime.h>
#include <hip/hip_bf16.h>
#include <stdint.h>

// ---------------------------------------------------------------------------
// EdgeFeatCat fused pipeline for gfx1250 (MI455X), bf16 WMMA path.
//   B=32, N=96, NODE_DIM=2048, COND_DIM=1024, E=128
//   edges = 32 * 96*95 = 291840, e_per_batch = 9120 (divisible by 16)
// ---------------------------------------------------------------------------

#define BDIM   32
#define NDIM   96
#define NODE_D 2048
#define COND_D 1024
#define EDIM   128
#define EDGES  291840
#define NN2    (NDIM * NDIM)          // 9216

typedef __attribute__((ext_vector_type(16))) __bf16 v16bf;
typedef __attribute__((ext_vector_type(8)))  float  v8f;
typedef __attribute__((ext_vector_type(4)))  unsigned int u32x4;
typedef __attribute__((ext_vector_type(4)))  float  f32x4;

union AB16 {
    v16bf v;
    unsigned short u[16];
    u32x4 q[2];
};

// fp32 -> bf16 via hardware cvt (RNE); backend packs pairs into v_cvt_pk_bf16_f32
__device__ __forceinline__ unsigned short f2bf(float f) {
    union { __bf16 h; unsigned short u; } c;
    c.h = (__bf16)f;
    return c.u;
}

__device__ __forceinline__ v8f wmma_bf16(v16bf a, v16bf b, v8f c) {
    // (neg_a, A, neg_b, B, c_mod, C, reuse_a, reuse_b)
    return __builtin_amdgcn_wmma_f32_16x16x32_bf16(false, a, false, b,
                                                   (short)0, c, false, false);
}

// B-fragment (32x16, bf16) from row-major W[out=128][ldK], tile column ntile,
// K block base kb32.  ISA layout: lanes 0-15 hold K {0..7, 16..23}, lanes 16-31
// hold K {8..15, 24..31} (matches the A-fragment layout).
__device__ __forceinline__ v16bf load_b_frag(const unsigned short* W, int ldK,
                                             int ntile, int kb32, int lane) {
    int n = (ntile << 4) + (lane & 15);
    int k = kb32 + ((lane & 16) ? 8 : 0);
    const u32x4* p = (const u32x4*)(W + n * ldK + k);
    AB16 f;
    f.q[0] = p[0];
    f.q[1] = p[2];
    return f.v;
}

// A-fragment (16x32, bf16) from a contiguous bf16 run `rowk` (>=32 elems).
// ISA layout: lane<16 -> K {0..7, 16..23}; lane>=16 -> K {8..15, 24..31}.
__device__ __forceinline__ v16bf load_a_frag_bf(const unsigned short* rowk, int lane) {
    int koff = (lane & 16) ? 8 : 0;
    AB16 f;
    f.q[0] = *(const u32x4*)(rowk + koff);
    f.q[1] = *(const u32x4*)(rowk + koff + 16);
    return f.v;
}

// Same A-fragment but sourced from fp32 with on-the-fly bf16 conversion.
__device__ __forceinline__ v16bf load_a_frag_f32(const float* rowk, int lane) {
    int koff = (lane & 16) ? 8 : 0;
    f32x4 c0 = *(const f32x4*)(rowk + koff);
    f32x4 c1 = *(const f32x4*)(rowk + koff + 4);
    f32x4 c2 = *(const f32x4*)(rowk + koff + 16);
    f32x4 c3 = *(const f32x4*)(rowk + koff + 20);
    v16bf r;
    r[0]  = (__bf16)c0.x; r[1]  = (__bf16)c0.y; r[2]  = (__bf16)c0.z; r[3]  = (__bf16)c0.w;
    r[4]  = (__bf16)c1.x; r[5]  = (__bf16)c1.y; r[6]  = (__bf16)c1.z; r[7]  = (__bf16)c1.w;
    r[8]  = (__bf16)c2.x; r[9]  = (__bf16)c2.y; r[10] = (__bf16)c2.z; r[11] = (__bf16)c2.w;
    r[12] = (__bf16)c3.x; r[13] = (__bf16)c3.y; r[14] = (__bf16)c3.z; r[15] = (__bf16)c3.w;
    return r;
}

// Async global->LDS b128 copy (CDNA5 VGLOBAL async path, tracked by ASYNCcnt).
__device__ __forceinline__ void async_copy_b128(unsigned lds_byte_off,
                                                const void* gsrc) {
    asm volatile("global_load_async_to_lds_b128 %0, %1, off"
                 :: "v"(lds_byte_off), "v"(gsrc)
                 : "memory");
}

__device__ __forceinline__ void wait_asynccnt0() {
    asm volatile("s_wait_asynccnt 0x0" ::: "memory");
}

// ---------------------------------------------------------------------------
// Kernel 1: weight-norm all matrices (W = v * g/||v||_row), emit bf16
// (q weights stay fp32 for the tiny fp32 GEMV).  512 blocks: mat = bid>>7.
// ---------------------------------------------------------------------------
__global__ void prep_weights(const float* node_v, const float* node_g,
                             const float* joint_v, const float* joint_g,
                             const float* lin_v, const float* lin_g,
                             const float* q_v, const float* q_g,
                             unsigned short* node_w, unsigned short* joint_w,
                             unsigned short* lin_w, float* q_w) {
    __shared__ float red[256];
    int mat = blockIdx.x >> 7;
    int row = blockIdx.x & 127;
    const float* v; const float* g; int K;
    if      (mat == 0) { v = node_v;  g = node_g;  K = NODE_D; }
    else if (mat == 1) { v = joint_v; g = joint_g; K = 2 * EDIM; }
    else if (mat == 2) { v = lin_v;   g = lin_g;   K = 2 * EDIM; }
    else               { v = q_v;     g = q_g;     K = COND_D; }

    const float* vr = v + (size_t)row * K;
    float s = 0.f;
    for (int k = threadIdx.x; k < K; k += 256) { float x = vr[k]; s += x * x; }
    red[threadIdx.x] = s;
    __syncthreads();
    for (int off = 128; off > 0; off >>= 1) {
        if ((int)threadIdx.x < off) red[threadIdx.x] += red[threadIdx.x + off];
        __syncthreads();
    }
    float scale = g[row] * __frsqrt_rn(red[0]);   // g / ||v||

    for (int k = threadIdx.x; k < K; k += 256) {
        float w = vr[k] * scale;
        if      (mat == 0) node_w [row * NODE_D    + k] = f2bf(w);
        else if (mat == 1) joint_w[row * (2*EDIM)  + k] = f2bf(w);
        else if (mat == 2) lin_w  [row * (2*EDIM)  + k] = f2bf(w);
        else               q_w    [row * COND_D    + k] = w;
    }
}

// ---------------------------------------------------------------------------
// Kernel 2: q = relu(cond @ q_W^T + q_b), stored bf16 [32,128].  Tiny.
// ---------------------------------------------------------------------------
__global__ void compute_q(const float* cond, const float* q_w, const float* q_b,
                          unsigned short* q_bf) {
    int b = blockIdx.x;
    int o = threadIdx.x;
    const float* c = cond + b * COND_D;
    const float* w = q_w + o * COND_D;
    float s = q_b[o];
    for (int k = 0; k < COND_D; ++k) s = fmaf(c[k], w[k], s);
    q_bf[b * EDIM + o] = f2bf(fmaxf(s, 0.f));
}

// ---------------------------------------------------------------------------
// Kernel 3: h = relu(X @ node_W^T + node_b), X=[3072,2048] fp32, H bf16.
// One 16x128 output tile per wave, K-loop of 64 x wmma 16x16x32.
// ---------------------------------------------------------------------------
__global__ void node_proj(const float* __restrict__ X,
                          const unsigned short* __restrict__ W,
                          const float* __restrict__ bias,
                          unsigned short* __restrict__ H) {
    int lane = threadIdx.x & 31;
    int wave = threadIdx.x >> 5;
    int tile = blockIdx.x * 8 + wave;          // 192 tiles total
    int m0 = tile << 4;
    int nl = lane & 15;
    int ms = (lane & 16) ? 8 : 0;

    const float* row = X + (size_t)(m0 + nl) * NODE_D;

    v8f z = {};
    v8f acc[8];
#pragma unroll
    for (int t = 0; t < 8; ++t) acc[t] = z;

    for (int kb = 0; kb < NODE_D; kb += 32) {
        v16bf a = load_a_frag_f32(row + kb, lane);
#pragma unroll
        for (int nt = 0; nt < 8; ++nt) {
            v16bf b = load_b_frag(W, NODE_D, nt, kb, lane);
            acc[nt] = wmma_bf16(a, b, acc[nt]);
        }
    }

    // C/D layout: lanes 0-15 hold (M=v, N=lane); lanes 16-31 hold (M=8+v, N=lane-16)
#pragma unroll
    for (int nt = 0; nt < 8; ++nt) {
        int n = (nt << 4) + nl;
        float bv = bias[n];
#pragma unroll
        for (int v = 0; v < 8; ++v) {
            float x = fmaxf(acc[nt][v] + bv, 0.f);
            H[(size_t)(m0 + ms + v) * EDIM + n] = f2bf(x);
        }
    }
}

// ---------------------------------------------------------------------------
// Kernel 4: fused edge pipeline.  One 16-edge tile per wave:
//   fusion = [h[i], h[j]] (K=256)  -> joint = relu(fusion @ Wj^T + bj)
//   cat    = [joint, q[b]] (K=256) -> out   = relu(cat @ Wl^T + bl)
// Both weight matrices are staged block-wide into LDS via the CDNA5 async
// global->LDS path; B fragments are then served by ds_load_b128.  The joint
// result is re-shaped D->A through a per-wave LDS tile between the chains.
// Dynamic LDS: [0,64K) jt reshape, [64K,128K) Wj, [128K,192K) Wl.
// ---------------------------------------------------------------------------
__global__ void edge_fused(const int* __restrict__ edge_idx,
                           const unsigned short* __restrict__ H,
                           const unsigned short* __restrict__ Wj,
                           const float* __restrict__ bj,
                           const unsigned short* __restrict__ Wl,
                           const float* __restrict__ bl,
                           const unsigned short* __restrict__ Qbf,
                           float* __restrict__ out) {
    extern __shared__ __align__(16) char smem[];
    unsigned short* jt  = (unsigned short*)smem;              // 8 x (16x128)
    unsigned short* sWj = (unsigned short*)(smem + (64 << 10));
    unsigned short* sWl = (unsigned short*)(smem + (128 << 10));

    int lane = threadIdx.x & 31;
    int wave = threadIdx.x >> 5;
    int tile = blockIdx.x * 8 + wave;          // 18240 tiles total
    int e0 = tile << 4;
    int nl = lane & 15;
    int ms = (lane & 16) ? 8 : 0;

    // ---- stage weights into LDS (async, 128KB, 512B per thread) ----
    {
        unsigned ldsWj = (unsigned)(uintptr_t)sWj;
        unsigned ldsWl = (unsigned)(uintptr_t)sWl;
#pragma unroll
        for (int it = 0; it < 16; ++it) {
            unsigned boff = (unsigned)(it * 256 + threadIdx.x) * 16u;
            async_copy_b128(ldsWj + boff, (const char*)Wj + boff);
            async_copy_b128(ldsWl + boff, (const char*)Wl + boff);
        }
        wait_asynccnt0();
    }
    __syncthreads();

    // decode this lane's A-row edge (all 16 edges of a tile share a batch)
    int idx = edge_idx[e0 + nl];
    int bi  = idx / NN2;
    int rem = idx - bi * NN2;
    int ii  = rem / NDIM;
    int jj  = rem - ii * NDIM;
    const unsigned short* rowI = H + (size_t)(bi * NDIM + ii) * EDIM;
    const unsigned short* rowJ = H + (size_t)(bi * NDIM + jj) * EDIM;

    v8f z = {};
    v8f acc[8];
#pragma unroll
    for (int t = 0; t < 8; ++t) acc[t] = z;

    // ---- stage 1: joint = fusion @ Wj^T ----
#pragma unroll
    for (int kb = 0; kb < 8; ++kb) {
        const unsigned short* src = (kb < 4) ? (rowI + kb * 32)
                                             : (rowJ + (kb - 4) * 32);
        v16bf a = load_a_frag_bf(src, lane);
#pragma unroll
        for (int nt = 0; nt < 8; ++nt) {
            v16bf b = load_b_frag(sWj, 2 * EDIM, nt, kb * 32, lane);
            acc[nt] = wmma_bf16(a, b, acc[nt]);
        }
    }

    // bias + relu, spill D-layout -> row-major bf16 tile in LDS
    unsigned short* my = jt + wave * 16 * EDIM;
#pragma unroll
    for (int nt = 0; nt < 8; ++nt) {
        int n = (nt << 4) + nl;
        float bv = bj[n];
#pragma unroll
        for (int v = 0; v < 8; ++v) {
            my[(ms + v) * EDIM + n] = f2bf(fmaxf(acc[nt][v] + bv, 0.f));
        }
    }
    __syncthreads();   // cross-lane LDS dependency within the wave

    // ---- stage 2: out = [joint, q] @ Wl^T ----
    const unsigned short* qrow = Qbf + bi * EDIM;
    v8f acc2[8];
#pragma unroll
    for (int t = 0; t < 8; ++t) acc2[t] = z;

#pragma unroll
    for (int kb = 0; kb < 8; ++kb) {
        v16bf a;
        if (kb < 4)
            a = load_a_frag_bf(my + nl * EDIM + kb * 32, lane);     // joint half (LDS)
        else
            a = load_a_frag_bf(qrow + (kb - 4) * 32, lane);         // q half (broadcast)
#pragma unroll
        for (int nt = 0; nt < 8; ++nt) {
            v16bf b = load_b_frag(sWl, 2 * EDIM, nt, kb * 32, lane);
            acc2[nt] = wmma_bf16(a, b, acc2[nt]);
        }
    }

    float* orow = out + (size_t)e0 * EDIM;
#pragma unroll
    for (int nt = 0; nt < 8; ++nt) {
        int n = (nt << 4) + nl;
        float bv = bl[n];
#pragma unroll
        for (int v = 0; v < 8; ++v) {
            orow[(size_t)(ms + v) * EDIM + n] = fmaxf(acc2[nt][v] + bv, 0.f);
        }
    }
}

// ---------------------------------------------------------------------------
extern "C" void kernel_launch(void* const* d_in, const int* in_sizes, int n_in,
                              void* d_out, int out_size, void* d_ws, size_t ws_size,
                              hipStream_t stream) {
    const float* node_feats = (const float*)d_in[0];
    const float* cond_feats = (const float*)d_in[1];
    const int*   edge_idx   = (const int*)  d_in[2];
    const float* node_v  = (const float*)d_in[3];
    const float* node_g  = (const float*)d_in[4];
    const float* node_b  = (const float*)d_in[5];
    const float* joint_v = (const float*)d_in[6];
    const float* joint_g = (const float*)d_in[7];
    const float* joint_b = (const float*)d_in[8];
    const float* q_v     = (const float*)d_in[9];
    const float* q_g     = (const float*)d_in[10];
    const float* q_b     = (const float*)d_in[11];
    const float* lin_v   = (const float*)d_in[12];
    const float* lin_g   = (const float*)d_in[13];
    const float* lin_b   = (const float*)d_in[14];

    // workspace layout (~1.93 MB)
    char* ws = (char*)d_ws;
    unsigned short* node_w  = (unsigned short*)ws; ws += (size_t)EDIM * NODE_D * 2;
    unsigned short* joint_w = (unsigned short*)ws; ws += (size_t)EDIM * 2 * EDIM * 2;
    unsigned short* lin_w   = (unsigned short*)ws; ws += (size_t)EDIM * 2 * EDIM * 2;
    float*          q_w     = (float*)ws;          ws += (size_t)EDIM * COND_D * 4;
    unsigned short* q_bf    = (unsigned short*)ws; ws += (size_t)BDIM * EDIM * 2;
    unsigned short* H       = (unsigned short*)ws; ws += (size_t)BDIM * NDIM * EDIM * 2;

    prep_weights<<<512, 256, 0, stream>>>(node_v, node_g, joint_v, joint_g,
                                          lin_v, lin_g, q_v, q_g,
                                          node_w, joint_w, lin_w, q_w);
    compute_q<<<BDIM, EDIM, 0, stream>>>(cond_feats, q_w, q_b, q_bf);
    node_proj<<<(BDIM * NDIM / 16) / 8, 256, 0, stream>>>(node_feats, node_w, node_b, H);
    edge_fused<<<(EDGES / 16) / 8, 256, 192 * 1024, stream>>>(edge_idx, H,
                                                              joint_w, joint_b,
                                                              lin_w, lin_b,
                                                              q_bf, (float*)d_out);
}
